// ManifoldProjection_24945170055751
// MI455X (gfx1250) — matrix operand order repaired and checked by reading, not compiled, not run on hardware
//
#include <hip/hip_runtime.h>
#include <hip/hip_bf16.h>

typedef __attribute__((ext_vector_type(16))) _Float16 v16h;
typedef __attribute__((ext_vector_type(8)))  _Float16 v8h;
typedef __attribute__((ext_vector_type(8)))  float    v8f;

#define T_Q     4096
#define N_DB    50000
#define N_DBP   50176           // 8 slices * 6272, multiple of 32
#define DIMS    512
#define K_NN    10
#define QT      128             // queries per workgroup
#define NCW     32              // db rows (B columns) per iteration
#define SLICES  8
#define SL_NIT  196             // iterations per slice: 196*32 = 6272 rows
#define BROW_H  520             // padded LDS row pitch in halves (1040 B, 16B aligned)
#define STILE_W 33              // padded score-tile pitch in floats

// ---------------------------------------------------------------------------
// Phase A: f32 -> f16 row conversion (+ optional squared-norm). 1 wave / row.
// ---------------------------------------------------------------------------
__global__ __launch_bounds__(256)
void convert_rows_kernel(const float* __restrict__ src, _Float16* __restrict__ dst,
                         float* __restrict__ norms, int nrows, int nvalid) {
    const int wave = threadIdx.x >> 5;
    const int lane = threadIdx.x & 31;
    const int row  = blockIdx.x * 8 + wave;
    if (row >= nrows) return;
    float ss = 0.0f;
    if (row < nvalid) {
        const float* sp = src + (size_t)row * DIMS;
        _Float16*    dp = dst + (size_t)row * DIMS;
        #pragma unroll
        for (int d0 = 0; d0 < DIMS / 32; ++d0) {
            const int d = d0 * 32 + lane;
            const float v = sp[d];
            dp[d] = (_Float16)v;
            ss += v * v;
        }
    } else {
        _Float16* dp = dst + (size_t)row * DIMS;
        #pragma unroll
        for (int d0 = 0; d0 < DIMS / 32; ++d0)
            dp[d0 * 32 + lane] = (_Float16)0.0f;
    }
    if (norms) {
        #pragma unroll
        for (int s = 16; s >= 1; s >>= 1)
            ss += __shfl_xor(ss, s, 32);
        if (lane == 0) norms[row] = ss;
    }
}

// ---------------------------------------------------------------------------
// async stage: copy NCW*1KB contiguous db rows into (row-padded) LDS buffer.
// Each wave copies 4 KB: 8 issues x 32 lanes x 16 B. Tracked by ASYNCcnt.
// ---------------------------------------------------------------------------
__device__ __forceinline__ void stage_async(const _Float16* dbh, int cbrow,
                                            uint32_t lds_base, int wave, int lane) {
    const uint64_t g0 = (uint64_t)(uintptr_t)dbh + (uint64_t)cbrow * (DIMS * 2);
    #pragma unroll
    for (int j = 0; j < 8; ++j) {
        const int flat   = wave * 4096 + j * 512 + lane * 16;   // byte in chunk
        const int row    = flat >> 10;                          // /1024
        const int within = flat & 1023;
        const uint32_t laddr = lds_base + (uint32_t)(row * (BROW_H * 2) + within);
        const uint64_t gaddr = g0 + (uint64_t)flat;
        asm volatile("global_load_async_to_lds_b128 %0, %1, off"
                     :: "v"(laddr), "v"(gaddr) : "memory");
    }
}

// ---------------------------------------------------------------------------
// Phase B: WMMA scores + per-query top-10 per DB slice.
// Grid (32, 8): x = 128-query tile, y = DB slice of 6272 rows.
// score s = x.db - 0.5*||db||^2   (max s == min distance)
// ---------------------------------------------------------------------------
__global__ __launch_bounds__(256)
void topk_kernel(const _Float16* __restrict__ xh, const _Float16* __restrict__ dbh,
                 const float* __restrict__ dbn,
                 float* __restrict__ cv, int* __restrict__ ci) {
    __shared__ _Float16 s_b[2 * NCW * BROW_H];       // 66.5 KB double-buffered B
    __shared__ float    s_tile[QT * STILE_W];        // 16.9 KB scores
    __shared__ float    s_mv[QT * 2 * K_NN];         // 10 KB
    __shared__ int      s_mi[QT * 2 * K_NN];         // 10 KB

    const int tid  = threadIdx.x;
    const int w    = tid >> 5;
    const int lane = tid & 31;
    const int lrow = lane & 15;
    const int lhi  = lane >> 4;
    const int qbase = blockIdx.x * QT;
    const int srow  = blockIdx.y * (SL_NIT * NCW);

    // Preload A fragments: this wave's 16 query rows, all 16 K-steps (128 VGPRs)
    const _Float16* aptr = xh + (size_t)(qbase + 16 * w + lrow) * DIMS + (lhi ? 16 : 0);
    v16h afr[16];
    #pragma unroll
    for (int ks = 0; ks < 16; ++ks) {
        v8h lo = *(const v8h*)(aptr + ks * 32);
        v8h hi = *(const v8h*)(aptr + ks * 32 + 8);
        afr[ks] = __builtin_shufflevector(lo, hi,
                    0, 1, 2, 3, 4, 5, 6, 7, 8, 9, 10, 11, 12, 13, 14, 15);
    }

    float tv[K_NN]; int ti[K_NN];
    #pragma unroll
    for (int i = 0; i < K_NN; ++i) { tv[i] = -3.0e38f; ti[i] = 0; }

    const uint32_t lb[2] = { (uint32_t)(uintptr_t)&s_b[0],
                             (uint32_t)(uintptr_t)&s_b[NCW * BROW_H] };

    // Prologue: stage chunk 0 into buffer 0
    stage_async(dbh, srow, lb[0], w, lane);
    asm volatile("s_wait_asynccnt 0" ::: "memory");
    __syncthreads();

    const int q_l  = tid >> 1;        // owned query (update phase)
    const int half = tid & 1;         // 16-column half

    for (int it = 0; it < SL_NIT; ++it) {
        const int cb   = srow + it * NCW;
        const int bufo = (it & 1) * (NCW * BROW_H);
        if (it + 1 < SL_NIT)
            stage_async(dbh, cb + NCW, lb[(it + 1) & 1], w, lane);

        v8f acc0 = {}, acc1 = {};
        #pragma unroll
        for (int ks = 0; ks < 16; ++ks) {
            const int boff = ks * 32 + lhi * 16;
            v8h b0l = *(const v8h*)&s_b[bufo + lrow * BROW_H + boff];
            v8h b0h = *(const v8h*)&s_b[bufo + lrow * BROW_H + boff + 8];
            v8h b1l = *(const v8h*)&s_b[bufo + (16 + lrow) * BROW_H + boff];
            v8h b1h = *(const v8h*)&s_b[bufo + (16 + lrow) * BROW_H + boff + 8];
            v16h b0 = __builtin_shufflevector(b0l, b0h,
                        0, 1, 2, 3, 4, 5, 6, 7, 8, 9, 10, 11, 12, 13, 14, 15);
            v16h b1 = __builtin_shufflevector(b1l, b1h,
                        0, 1, 2, 3, 4, 5, 6, 7, 8, 9, 10, 11, 12, 13, 14, 15);
            acc0 = __builtin_amdgcn_wmma_f32_16x16x32_f16(
                       false, afr[ks], false, b0, (short)0, acc0, false, false);
            acc1 = __builtin_amdgcn_wmma_f32_16x16x32_f16(
                       false, afr[ks], false, b1, (short)0, acc1, false, false);
        }

        asm volatile("s_wait_asynccnt 0" ::: "memory");
        __syncthreads();   // barrier A: prev topk done with s_tile; next B buf ready

        const int n0 = cb + lrow, n1 = cb + 16 + lrow;
        const float d0 = (n0 < N_DB) ? dbn[n0] : 0.0f;
        const float d1 = (n1 < N_DB) ? dbn[n1] : 0.0f;
        #pragma unroll
        for (int r = 0; r < 8; ++r) {
            const int ql = 16 * w + 8 * lhi + r;
            s_tile[ql * STILE_W + lrow]      = (n0 < N_DB) ? (acc0[r] - 0.5f * d0) : -3.0e38f;
            s_tile[ql * STILE_W + 16 + lrow] = (n1 < N_DB) ? (acc1[r] - 0.5f * d1) : -3.0e38f;
        }
        __syncthreads();   // barrier B: scores visible

        #pragma unroll
        for (int j = 0; j < 16; ++j) {
            const int   c2  = half * 16 + j;
            const float val = s_tile[q_l * STILE_W + c2];
            if (val > tv[0]) {
                const int idx = cb + c2;
                int p = 0;
                #pragma unroll
                for (int i = 1; i < K_NN; ++i) {
                    if (val > tv[i]) { tv[i - 1] = tv[i]; ti[i - 1] = ti[i]; p = i; }
                }
                tv[p] = val; ti[p] = idx;
            }
        }
    }

    // merge the 2 per-query lists -> slice top-10 candidates (best-first)
    #pragma unroll
    for (int i = 0; i < K_NN; ++i) {
        s_mv[(q_l * 2 + half) * K_NN + i] = tv[i];
        s_mi[(q_l * 2 + half) * K_NN + i] = ti[i];
    }
    __syncthreads();
    if (tid < QT) {
        const int q = tid;
        const size_t base = ((size_t)(qbase + q) * SLICES + blockIdx.y) * K_NN;
        for (int r = 0; r < K_NN; ++r) {
            float best = -3.2e38f; int bi = 0, bp = 0;
            for (int e = 0; e < 2 * K_NN; ++e) {
                const float v = s_mv[q * 2 * K_NN + e];
                if (v > best) { best = v; bi = s_mi[q * 2 * K_NN + e]; bp = e; }
            }
            s_mv[q * 2 * K_NN + bp] = -3.3e38f;
            cv[base + r] = best;
            ci[base + r] = bi;
        }
    }
}

// ---------------------------------------------------------------------------
// Phase C: 8-way candidate merge (register-only wave argmax) + LLE solve +
// fusion. 1 wave per query, 8 waves per block.
// ---------------------------------------------------------------------------
__global__ __launch_bounds__(256)
void lle_kernel(const float* __restrict__ x, const float* __restrict__ db,
                const float* __restrict__ cv, const int* __restrict__ ci,
                float* __restrict__ out) {
    const int wave = threadIdx.x >> 5;
    const int lane = threadIdx.x & 31;
    const int t    = blockIdx.x * 8 + wave;
    if (t >= T_Q) return;

    // ---- merge 80 candidates -> top-10 global neighbors -------------------
    const size_t cb = (size_t)t * (SLICES * K_NN);   // 80 entries
    float v0 = cv[cb + lane];
    float v1 = cv[cb + lane + 32];
    float v2 = (lane < 16) ? cv[cb + lane + 64] : -3.4e38f;
    int   i0 = ci[cb + lane];
    int   i1 = ci[cb + lane + 32];
    int   i2 = (lane < 16) ? ci[cb + lane + 64] : 0;

    int idx[K_NN];
    #pragma unroll
    for (int r = 0; r < K_NN; ++r) {
        float mv = v0; int mi = i0; int ms = 0;
        if (v1 > mv) { mv = v1; mi = i1; ms = 1; }
        if (v2 > mv) { mv = v2; mi = i2; ms = 2; }
        float bv = mv; int bi = mi; int bl = lane;
        #pragma unroll
        for (int s = 16; s >= 1; s >>= 1) {
            const float ov = __shfl_xor(bv, s, 32);
            const int   oi = __shfl_xor(bi, s, 32);
            const int   ol = __shfl_xor(bl, s, 32);
            if (ov > bv || (ov == bv && ol < bl)) { bv = ov; bi = oi; bl = ol; }
        }
        idx[r] = bi;
        if (lane == bl) {
            if (ms == 0)      v0 = -3.4e38f;
            else if (ms == 1) v1 = -3.4e38f;
            else              v2 = -3.4e38f;
        }
    }

    // ---- LLE: ATA (45 sym) + ATB over 512 dims, wave-reduced --------------
    float ata[45], atb[9];
    #pragma unroll
    for (int c = 0; c < 45; ++c) ata[c] = 0.0f;
    #pragma unroll
    for (int j = 0; j < 9; ++j) atb[j] = 0.0f;

    const float* f0p = db + (size_t)idx[0] * DIMS;
    const float* xp  = x  + (size_t)t * DIMS;

    for (int d0 = 0; d0 < DIMS / 32; ++d0) {
        const int d = d0 * 32 + lane;
        const float f0v = f0p[d];
        const float b   = xp[d] - f0v;
        float a[9];
        #pragma unroll
        for (int k = 0; k < 9; ++k) a[k] = db[(size_t)idx[k + 1] * DIMS + d] - f0v;
        int c = 0;
        #pragma unroll
        for (int j = 0; j < 9; ++j) {
            atb[j] += a[j] * b;
            #pragma unroll
            for (int k = j; k < 9; ++k) ata[c++] += a[j] * a[k];
        }
    }
    #pragma unroll
    for (int s = 16; s >= 1; s >>= 1) {
        #pragma unroll
        for (int c = 0; c < 45; ++c) ata[c] += __shfl_xor(ata[c], s, 32);
        #pragma unroll
        for (int j = 0; j < 9; ++j)  atb[j] += __shfl_xor(atb[j], s, 32);
    }

    float M[9][10];
    #pragma unroll
    for (int j = 0; j < 9; ++j) {
        #pragma unroll
        for (int k = 0; k < 9; ++k) {
            const int jj = j < k ? j : k, kk = j < k ? k : j;
            M[j][k] = ata[jj * 9 - jj * (jj - 1) / 2 + (kk - jj)];
        }
        M[j][9] = atb[j];
    }
    #pragma unroll
    for (int col = 0; col < 9; ++col) {
        const float inv = 1.0f / M[col][col];
        #pragma unroll
        for (int r2 = col + 1; r2 < 9; ++r2) {
            const float f = M[r2][col] * inv;
            #pragma unroll
            for (int c2 = col; c2 < 10; ++c2) M[r2][c2] -= f * M[col][c2];
        }
    }
    float wgt[K_NN];
    #pragma unroll
    for (int r2 = 8; r2 >= 0; --r2) {
        float s = M[r2][9];
        #pragma unroll
        for (int c2 = r2 + 1; c2 < 9; ++c2) s -= M[r2][c2] * wgt[c2 + 1];
        wgt[r2 + 1] = s / M[r2][r2];
    }
    float sum = 0.0f;
    #pragma unroll
    for (int k = 1; k < K_NN; ++k) sum += wgt[k];
    wgt[0] = 1.0f - sum;

    for (int d0 = 0; d0 < DIMS / 32; ++d0) {
        const int d = d0 * 32 + lane;
        float fuse = 0.0f;
        #pragma unroll
        for (int k = 0; k < K_NN; ++k) fuse += wgt[k] * db[(size_t)idx[k] * DIMS + d];
        out[(size_t)t * DIMS + d] = xp[d] * 0.2f + fuse * 0.8f;
    }
}

// ---------------------------------------------------------------------------
extern "C" void kernel_launch(void* const* d_in, const int* in_sizes, int n_in,
                              void* d_out, int out_size, void* d_ws, size_t ws_size,
                              hipStream_t stream) {
    (void)in_sizes; (void)n_in; (void)out_size; (void)ws_size;
    const float* x  = (const float*)d_in[0];   // (4096, 512)
    const float* db = (const float*)d_in[1];   // (50000, 512)
    float* out = (float*)d_out;

    char* ws = (char*)d_ws;
    size_t off = 0;
    _Float16* dbh = (_Float16*)(ws + off); off += (size_t)N_DBP * DIMS * sizeof(_Float16);
    _Float16* xh  = (_Float16*)(ws + off); off += (size_t)T_Q   * DIMS * sizeof(_Float16);
    float*    dbn = (float*)   (ws + off); off += (size_t)N_DBP * sizeof(float);
    float*    cv  = (float*)   (ws + off); off += (size_t)T_Q * SLICES * K_NN * sizeof(float);
    int*      ci  = (int*)     (ws + off);

    convert_rows_kernel<<<N_DBP / 8, 256, 0, stream>>>(db, dbh, dbn, N_DBP, N_DB);
    convert_rows_kernel<<<T_Q / 8,   256, 0, stream>>>(x,  xh,  nullptr, T_Q, T_Q);
    dim3 tg(T_Q / QT, SLICES);
    topk_kernel<<<tg, 256, 0, stream>>>(xh, dbh, dbn, cv, ci);
    lle_kernel<<<T_Q / 8, 256, 0, stream>>>(x, db, cv, ci, out);
}
